// Fast_RCNN_5506148074060
// MI455X (gfx1250) — compile-verified, matrix operand order checked by reading, not run on hardware
//
#include <hip/hip_runtime.h>
#include <hip/hip_bf16.h>

typedef __attribute__((ext_vector_type(16))) _Float16 v16h;
typedef __attribute__((ext_vector_type(8)))  float    v8f;
typedef __attribute__((ext_vector_type(4)))  unsigned int v4u;

#define BATCH 8
#define RPI   128            // ROIs per image
#define NROI  (BATCH * RPI)  // 1024
#define FDIM  12544          // 256*7*7
#define HDIM  4096

#define LDS_STRIDE 40        // f16 elems per LDS row (32 data + 8 pad) -> 80B,
                             // makes 16-lane ds_load_b128 bank-conflict-free

// ---------------------------------------------------------------------------
// Direct fp32 conv + ReLU (convs are ~7% of FLOPs; WMMA not worth it here)
// ---------------------------------------------------------------------------
__global__ void conv_relu_kernel(const float* __restrict__ in,
                                 const float* __restrict__ w,
                                 const float* __restrict__ bias,
                                 float* __restrict__ out,
                                 int Cin, int Hin, int Win,
                                 int Cout, int Hout, int Wout,
                                 int KS, int stride, int pad) {
    long long idx = (long long)blockIdx.x * blockDim.x + threadIdx.x;
    long long total = (long long)BATCH * Cout * Hout * Wout;
    if (idx >= total) return;
    int wo = idx % Wout;
    int ho = (idx / Wout) % Hout;
    int co = (idx / ((long long)Wout * Hout)) % Cout;
    int b  = idx / ((long long)Wout * Hout * Cout);

    float s = bias[co];
    const float* inb = in + (long long)b * Cin * Hin * Win;
    const float* wc  = w  + (long long)co * Cin * KS * KS;
    for (int ci = 0; ci < Cin; ++ci) {
        const float* inc = inb + (long long)ci * Hin * Win;
        const float* wk  = wc + ci * KS * KS;
        for (int kh = 0; kh < KS; ++kh) {
            int hi = ho * stride - pad + kh;
            if (hi < 0 || hi >= Hin) continue;
            for (int kw = 0; kw < KS; ++kw) {
                int wi = wo * stride - pad + kw;
                if (wi < 0 || wi >= Win) continue;
                s += inc[hi * Win + wi] * wk[kh * KS + kw];
            }
        }
    }
    out[idx] = fmaxf(s, 0.0f);
}

// ---------------------------------------------------------------------------
// 3x3 stride-2 VALID max pool
// ---------------------------------------------------------------------------
__global__ void maxpool3x3s2_kernel(const float* __restrict__ in, float* __restrict__ out,
                                    int C, int Hin, int Win, int Hout, int Wout) {
    long long idx = (long long)blockIdx.x * blockDim.x + threadIdx.x;
    long long total = (long long)BATCH * C * Hout * Wout;
    if (idx >= total) return;
    int wo = idx % Wout;
    int ho = (idx / Wout) % Hout;
    int c  = (idx / ((long long)Wout * Hout)) % C;
    int b  = idx / ((long long)Wout * Hout * C);
    const float* p = in + (((long long)b * C + c) * Hin) * Win;
    float m = -3.402823e38f;
    int h0 = ho * 2, w0 = wo * 2;
    for (int dh = 0; dh < 3; ++dh)
        for (int dw = 0; dw < 3; ++dw)
            m = fmaxf(m, p[(h0 + dh) * Win + (w0 + dw)]);
    out[idx] = m;
}

// ---------------------------------------------------------------------------
// torchvision roi_pool math; writes f16 directly into the FC1 A-matrix
// row layout = c*49 + ph*7 + pw  (matches reference reshape)
// ---------------------------------------------------------------------------
__global__ void roi_pool_kernel(const float* __restrict__ feat,   // [8,256,6,6]
                                const float* __restrict__ boxes,  // [8,128,4]
                                _Float16* __restrict__ Aout) {    // [1024,12544]
    const float SCALE = 13.0f / 227.0f;
    long long idx = (long long)blockIdx.x * blockDim.x + threadIdx.x;
    long long total = (long long)NROI * 256 * 49;
    if (idx >= total) return;
    int pw = idx % 7;
    int ph = (idx / 7) % 7;
    int c  = (idx / 49) % 256;
    int r  = (idx / (49 * 256)) % RPI;
    int b  = idx / ((long long)49 * 256 * RPI);

    const float* box = boxes + ((long long)b * RPI + r) * 4;
    float x1 = rintf(box[0] * SCALE), y1 = rintf(box[1] * SCALE);
    float x2 = rintf(box[2] * SCALE), y2 = rintf(box[3] * SCALE);
    float roi_w = fmaxf(x2 - x1 + 1.0f, 1.0f);
    float roi_h = fmaxf(y2 - y1 + 1.0f, 1.0f);
    float bh = roi_h / 7.0f, bw = roi_w / 7.0f;
    float hs = fminf(fmaxf(floorf(ph * bh) + y1, 0.0f), 6.0f);
    float he = fminf(fmaxf(ceilf((ph + 1) * bh) + y1, 0.0f), 6.0f);
    float ws = fminf(fmaxf(floorf(pw * bw) + x1, 0.0f), 6.0f);
    float we = fminf(fmaxf(ceilf((pw + 1) * bw) + x1, 0.0f), 6.0f);
    bool empty = (he <= hs) || (we <= ws);

    const float* f = feat + (((long long)b * 256 + c) * 6) * 6;
    float m = -1e30f;
    for (int h = 0; h < 6; ++h) {
        if ((float)h >= hs && (float)h < he)
            for (int w = 0; w < 6; ++w)
                if ((float)w >= ws && (float)w < we)
                    m = fmaxf(m, f[h * 6 + w]);
    }
    float val = empty ? 0.0f : m;
    Aout[((long long)b * RPI + r) * FDIM + (long long)c * 49 + ph * 7 + pw] = (_Float16)val;
}

// ---------------------------------------------------------------------------
// fp32 [K,N] -> f16 [N,K]  (weight transpose+downconvert so B fragments are
// contiguous 16B loads per the CDNA5 WMMA lane layout)
// ---------------------------------------------------------------------------
__global__ void transpose_f16_kernel(const float* __restrict__ W, _Float16* __restrict__ Bt,
                                     int K, int N) {
    long long i = (long long)blockIdx.x * blockDim.x + threadIdx.x;
    if (i >= (long long)K * N) return;
    int k = i % K;
    int n = i / K;
    Bt[(long long)n * K + k] = (_Float16)W[(long long)k * N + n];
}

// ---------------------------------------------------------------------------
// WMMA GEMM: C[M,N] = A[M,K](f16,row-major) * Bt[N,K](f16) + bias[N]
// wave32; 16x16x32 f16 WMMA; each wave 2x2 tiles (32x32); 8 waves -> 64x128.
// Tiles staged through LDS (padded rows, conflict-free ds_load_b128) with a
// register-mediated double buffer so global fetch overlaps WMMA issue.
// ---------------------------------------------------------------------------
__device__ __forceinline__ v16h lds_frag(const _Float16* p) {
    const v4u* q = (const v4u*)p;
    union { v16h h; v4u u[2]; } f;
    f.u[0] = q[0];   // K = k0 + hf*8 + 0..7
    f.u[1] = q[2];   // K = k0 + 16 + hf*8 + 0..7
    return f.h;
}

__device__ __forceinline__ v8f wmma_f16(v16h a, v16h b, v8f c) {
    return __builtin_amdgcn_wmma_f32_16x16x32_f16(false, a, false, b, (short)0, c, false, false);
}

__global__ __launch_bounds__(256)
void gemm_wmma_kernel(const _Float16* __restrict__ A,
                      const _Float16* __restrict__ Bt,
                      const float* __restrict__ bias,
                      float* __restrict__ C,        // optional f32 out
                      _Float16* __restrict__ Ch,    // optional f16 out
                      int M, int N, int K) {
    __shared__ __align__(16) _Float16 As[2][64 * LDS_STRIDE];
    __shared__ __align__(16) _Float16 Bs[2][128 * LDS_STRIDE];

    const int t    = threadIdx.x;
    const int lane = t & 31;
    const int wave = t >> 5;
    const int wm = wave >> 2;            // 0..1
    const int wn = wave & 3;             // 0..3
    const int mBase = blockIdx.y * 64;
    const int nBase = blockIdx.x * 128;
    const int hf  = (lane >> 4) & 1;
    const int l16 = lane & 15;

    const _Float16* Ag  = A  + (long long)mBase * K;
    const _Float16* Btg = Bt + (long long)nBase * K;

    // cooperative staging coordinates: A = 64 rows x 4 chunks (1/thread),
    // B = 128 rows x 4 chunks (2/thread)
    const int ar = t >> 2, ac = t & 3;       // A row, 8-f16 chunk
    const int br = t >> 2, bc = t & 3;       // B rows br and br+64

    const _Float16* gA  = Ag  + (long long)ar * K + ac * 8;
    const _Float16* gB0 = Btg + (long long)br * K + bc * 8;
    const _Float16* gB1 = Btg + (long long)(br + 64) * K + bc * 8;

    _Float16* sA  = &As[0][ar * LDS_STRIDE + ac * 8];
    _Float16* sB0 = &Bs[0][br * LDS_STRIDE + bc * 8];
    _Float16* sB1 = &Bs[0][(br + 64) * LDS_STRIDE + bc * 8];
    const int bufStrideA = 64 * LDS_STRIDE;
    const int bufStrideB = 128 * LDS_STRIDE;

    // preload first k-tile straight into LDS buffer 0
    *(v4u*)sA  = *(const v4u*)gA;
    *(v4u*)sB0 = *(const v4u*)gB0;
    *(v4u*)sB1 = *(const v4u*)gB1;
    __syncthreads();

    // per-wave fragment row offsets inside the LDS tiles
    const int aOff0 = (wm * 32 + l16)      * LDS_STRIDE + hf * 8;
    const int aOff1 = (wm * 32 + 16 + l16) * LDS_STRIDE + hf * 8;
    const int bOff0 = (wn * 32 + l16)      * LDS_STRIDE + hf * 8;
    const int bOff1 = (wn * 32 + 16 + l16) * LDS_STRIDE + hf * 8;

    v8f acc00 = {}, acc01 = {}, acc10 = {}, acc11 = {};
    int buf = 0;

    for (int k0 = 0; k0 < K; k0 += 32) {
        const bool nxt = (k0 + 32) < K;
        v4u ra, rb0, rb1;
        if (nxt) {
            // fetch next tile into registers while WMMAs run on LDS[buf]
            ra  = *(const v4u*)(gA  + k0 + 32);
            rb0 = *(const v4u*)(gB0 + k0 + 32);
            rb1 = *(const v4u*)(gB1 + k0 + 32);
            // and pull the tile after that toward L2 (global_prefetch_b8)
            __builtin_prefetch(gA  + k0 + 64, 0, 1);
            __builtin_prefetch(gB0 + k0 + 64, 0, 1);
        }

        const _Float16* as = As[buf];
        const _Float16* bs = Bs[buf];
        v16h a0 = lds_frag(as + aOff0);
        v16h a1 = lds_frag(as + aOff1);
        v16h b0 = lds_frag(bs + bOff0);
        v16h b1 = lds_frag(bs + bOff1);

        acc00 = wmma_f16(a0, b0, acc00);
        acc01 = wmma_f16(a0, b1, acc01);
        acc10 = wmma_f16(a1, b0, acc10);
        acc11 = wmma_f16(a1, b1, acc11);

        __syncthreads();
        if (nxt) {
            int o = (buf ^ 1) ? 1 : 0;
            *(v4u*)(sA  + o * bufStrideA) = ra;
            *(v4u*)(sB0 + o * bufStrideB) = rb0;
            *(v4u*)(sB1 + o * bufStrideB) = rb1;
        }
        __syncthreads();
        buf ^= 1;
    }

    // C/D layout: VGPR v -> M = v + hf*8 (per 16-row tile); lane%16 -> N
    v8f accs[4] = {acc00, acc01, acc10, acc11};
    for (int tt = 0; tt < 4; ++tt) {
        int tm = tt >> 1, tn = tt & 1;
        int n = nBase + wn * 32 + tn * 16 + l16;
        float bv = bias ? bias[n] : 0.0f;
        #pragma unroll
        for (int v = 0; v < 8; ++v) {
            int m = mBase + wm * 32 + tm * 16 + hf * 8 + v;
            float val = accs[tt][v] + bv;
            if (C)  C [(long long)m * N + n] = val;
            if (Ch) Ch[(long long)m * N + n] = (_Float16)val;
        }
    }
}

// ---------------------------------------------------------------------------
// Heads: cls [1024,2] and bbs [1024,8] from X [1024,4096]. 10240 dots, tiny.
// out = [cls flat (2048) | bbs flat (8192)]
// ---------------------------------------------------------------------------
__global__ void heads_kernel(const float* __restrict__ X,
                             const float* __restrict__ cw, const float* __restrict__ cb,
                             const float* __restrict__ bw, const float* __restrict__ bb,
                             float* __restrict__ out) {
    int i = blockIdx.x * blockDim.x + threadIdx.x;
    if (i >= NROI * 10) return;
    int row = i / 10;
    int j = i % 10;
    const float* x = X + (long long)row * HDIM;
    if (j < 2) {
        float s = cb[j];
        for (int k = 0; k < HDIM; ++k) s += x[k] * cw[k * 2 + j];
        out[row * 2 + j] = s;
    } else {
        int jj = j - 2;
        float s = bb[jj];
        for (int k = 0; k < HDIM; ++k) s += x[k] * bw[k * 8 + jj];
        out[NROI * 2 + row * 8 + jj] = s;
    }
}

// ---------------------------------------------------------------------------
extern "C" void kernel_launch(void* const* d_in, const int* in_sizes, int n_in,
                              void* d_out, int out_size, void* d_ws, size_t ws_size,
                              hipStream_t stream) {
    (void)in_sizes; (void)n_in; (void)out_size; (void)ws_size;
    const float* X     = (const float*)d_in[0];
    const float* boxes = (const float*)d_in[1];
    const float* w1 = (const float*)d_in[2];  const float* b1 = (const float*)d_in[3];
    const float* w2 = (const float*)d_in[4];  const float* b2 = (const float*)d_in[5];
    const float* w3 = (const float*)d_in[6];  const float* b3 = (const float*)d_in[7];
    const float* w4 = (const float*)d_in[8];  const float* b4 = (const float*)d_in[9];
    const float* w5 = (const float*)d_in[10]; const float* b5 = (const float*)d_in[11];
    const float* fc1_w = (const float*)d_in[12]; const float* fc1_b = (const float*)d_in[13];
    const float* fc2_w = (const float*)d_in[14]; const float* fc2_b = (const float*)d_in[15];
    const float* cls_w = (const float*)d_in[16]; const float* cls_b = (const float*)d_in[17];
    const float* bbs_w = (const float*)d_in[18]; const float* bbs_b = (const float*)d_in[19];
    float* out = (float*)d_out;

    char* ws = (char*)d_ws;
    size_t off = 0;
    auto alloc = [&](size_t bytes) -> void* {
        off = (off + 255) & ~(size_t)255;
        void* p = ws + off;
        off += bytes;
        return p;
    };

    float* conv1 = (float*)alloc((size_t)BATCH * 64  * 56 * 56 * 4);
    float* pool1 = (float*)alloc((size_t)BATCH * 64  * 27 * 27 * 4);
    float* conv2 = (float*)alloc((size_t)BATCH * 192 * 27 * 27 * 4);
    float* pool2 = (float*)alloc((size_t)BATCH * 192 * 13 * 13 * 4);
    float* conv3 = (float*)alloc((size_t)BATCH * 384 * 13 * 13 * 4);
    float* conv4 = (float*)alloc((size_t)BATCH * 256 * 13 * 13 * 4);
    float* conv5 = (float*)alloc((size_t)BATCH * 256 * 13 * 13 * 4);
    float* feats = (float*)alloc((size_t)BATCH * 256 * 6  * 6  * 4);
    _Float16* A1  = (_Float16*)alloc((size_t)NROI * FDIM * 2);
    _Float16* W1t = (_Float16*)alloc((size_t)HDIM * FDIM * 2);
    _Float16* A2  = (_Float16*)alloc((size_t)NROI * HDIM * 2);
    _Float16* W2t = (_Float16*)alloc((size_t)HDIM * HDIM * 2);
    float*    X2  = (float*)   alloc((size_t)NROI * HDIM * 4);

    const int TB = 256;
    auto blocks = [](long long n, int tb) { return (unsigned)((n + tb - 1) / tb); };

    // ---- AlexNet features ----
    conv_relu_kernel<<<blocks((long long)BATCH*64*56*56, TB), TB, 0, stream>>>(
        X, w1, b1, conv1, 3, 227, 227, 64, 56, 56, 11, 4, 2);
    maxpool3x3s2_kernel<<<blocks((long long)BATCH*64*27*27, TB), TB, 0, stream>>>(
        conv1, pool1, 64, 56, 56, 27, 27);
    conv_relu_kernel<<<blocks((long long)BATCH*192*27*27, TB), TB, 0, stream>>>(
        pool1, w2, b2, conv2, 64, 27, 27, 192, 27, 27, 5, 1, 2);
    maxpool3x3s2_kernel<<<blocks((long long)BATCH*192*13*13, TB), TB, 0, stream>>>(
        conv2, pool2, 192, 27, 27, 13, 13);
    conv_relu_kernel<<<blocks((long long)BATCH*384*13*13, TB), TB, 0, stream>>>(
        pool2, w3, b3, conv3, 192, 13, 13, 384, 13, 13, 3, 1, 1);
    conv_relu_kernel<<<blocks((long long)BATCH*256*13*13, TB), TB, 0, stream>>>(
        conv3, w4, b4, conv4, 384, 13, 13, 256, 13, 13, 3, 1, 1);
    conv_relu_kernel<<<blocks((long long)BATCH*256*13*13, TB), TB, 0, stream>>>(
        conv4, w5, b5, conv5, 256, 13, 13, 256, 13, 13, 3, 1, 1);
    maxpool3x3s2_kernel<<<blocks((long long)BATCH*256*6*6, TB), TB, 0, stream>>>(
        conv5, feats, 256, 13, 13, 6, 6);

    // ---- ROI pool -> f16 A-matrix for FC1 ----
    roi_pool_kernel<<<blocks((long long)NROI*256*49, TB), TB, 0, stream>>>(feats, boxes, A1);

    // ---- FC1 (dominant GEMM): 1024x4096x12544, WMMA f16 ----
    transpose_f16_kernel<<<blocks((long long)FDIM*HDIM, TB), TB, 0, stream>>>(fc1_w, W1t, FDIM, HDIM);
    {
        dim3 grid(HDIM / 128, NROI / 64);
        gemm_wmma_kernel<<<grid, 256, 0, stream>>>(A1, W1t, fc1_b, nullptr, A2,
                                                   NROI, HDIM, FDIM);
    }

    // ---- FC2: 1024x4096x4096, WMMA f16 ----
    transpose_f16_kernel<<<blocks((long long)HDIM*HDIM, TB), TB, 0, stream>>>(fc2_w, W2t, HDIM, HDIM);
    {
        dim3 grid(HDIM / 128, NROI / 64);
        gemm_wmma_kernel<<<grid, 256, 0, stream>>>(A2, W2t, fc2_b, X2, nullptr,
                                                   NROI, HDIM, HDIM);
    }

    // ---- Heads ----
    heads_kernel<<<blocks((long long)NROI*10, TB), TB, 0, stream>>>(
        X2, cls_w, cls_b, bbs_w, bbs_b, out);
}
